// Block_20684562497857
// MI455X (gfx1250) — matrix operand order repaired and checked by reading, not compiled, not run on hardware
//
#include <hip/hip_runtime.h>

#define B_   2
#define T_   2048
#define C_   1024
#define H_   16
#define HS_  64
#define NTOK (B_ * T_)

typedef __attribute__((ext_vector_type(16))) __bf16 bf16x16;
typedef __attribute__((ext_vector_type(8)))  __bf16 bf16x8;
typedef __attribute__((ext_vector_type(8)))  float  f32x8;

typedef unsigned short u16;
typedef unsigned int   u32;

// float -> bf16 (round to nearest even), stored as raw u16
static __device__ __forceinline__ u16 f2bf(float f) {
    u32 u = __float_as_uint(f);
    u32 r = u + 0x7fffu + ((u >> 16) & 1u);
    return (u16)(r >> 16);
}

union ABFrag { bf16x16 v; bf16x8 h[2]; };

// ---------------------------------------------------------------------------
// Weight preparation (all weights stored TRANSPOSED as bf16 [N][K])
// ---------------------------------------------------------------------------

// Wq/Wk/Wv (H, C, HS) fp32 -> bf16 [H*HS][C]   (i.e. output-major, K=C inner)
__global__ void wqkvT_kernel(const float* __restrict__ w, u16* __restrict__ out) {
    int i = blockIdx.x * blockDim.x + threadIdx.x;
    if (i >= H_ * C_ * HS_) return;
    int d = i % HS_;
    int c = (i / HS_) % C_;
    int h = i / (C_ * HS_);
    out[(size_t)(h * HS_ + d) * C_ + c] = f2bf(w[i]);
}

// W [K,N] fp32 -> bf16 [N,K]
__global__ void cvtT_kernel(const float* __restrict__ in, u16* __restrict__ out,
                            int K, int N) {
    int i = blockIdx.x * blockDim.x + threadIdx.x;
    if (i >= K * N) return;
    int k = i / N, n = i % N;
    out[(size_t)n * K + k] = f2bf(in[i]);
}

// v [B,T,H*HS] bf16 -> vT [B,H,HS,T] bf16
__global__ void vtrans_kernel(const u16* __restrict__ v, u16* __restrict__ vT) {
    int i = blockIdx.x * blockDim.x + threadIdx.x;
    if (i >= B_ * T_ * C_) return;
    int d = i % HS_;
    int h = (i / HS_) % H_;
    int t = (i / C_) % T_;
    int b = i / (T_ * C_);
    vT[((size_t)(b * H_ + h) * HS_ + d) * T_ + t] = v[i];
}

// ---------------------------------------------------------------------------
// LayerNorm: one 256-thread block per token row (C=1024 -> 4 elems/thread)
// ---------------------------------------------------------------------------
__global__ __launch_bounds__(256)
void ln_kernel(const float* __restrict__ x, const float* __restrict__ g,
               const float* __restrict__ bb, u16* __restrict__ out) {
    __shared__ float red[256];
    int row = blockIdx.x, tid = threadIdx.x;
    const float* xr = x + (size_t)row * C_;
    float4 v = ((const float4*)xr)[tid];
    red[tid] = v.x + v.y + v.z + v.w;
    __syncthreads();
    for (int o = 128; o > 0; o >>= 1) { if (tid < o) red[tid] += red[tid + o]; __syncthreads(); }
    float mu = red[0] * (1.0f / C_);
    __syncthreads();
    float dx = v.x - mu, dy = v.y - mu, dz = v.z - mu, dw = v.w - mu;
    red[tid] = dx * dx + dy * dy + dz * dz + dw * dw;
    __syncthreads();
    for (int o = 128; o > 0; o >>= 1) { if (tid < o) red[tid] += red[tid + o]; __syncthreads(); }
    float rstd = rsqrtf(red[0] * (1.0f / C_) + 1e-5f);
    int c0 = tid * 4;
    u16* orow = out + (size_t)row * C_;
    orow[c0 + 0] = f2bf(dx * rstd * g[c0 + 0] + bb[c0 + 0]);
    orow[c0 + 1] = f2bf(dy * rstd * g[c0 + 1] + bb[c0 + 1]);
    orow[c0 + 2] = f2bf(dz * rstd * g[c0 + 2] + bb[c0 + 2]);
    orow[c0 + 3] = f2bf(dw * rstd * g[c0 + 3] + bb[c0 + 3]);
}

// ---------------------------------------------------------------------------
// bf16 WMMA GEMM: out = A[M,K] @ B[K,N] (+bias)(+resid)(relu), B given as BT[N,K].
// 128x128x32 tiles, 256 threads = 8 waves (4 row-bands x 2 col-bands),
// each wave computes 32x64 = 8 WMMAs/k-step. Double-buffered LDS, one
// barrier per k-step, global_prefetch two tiles ahead.
// ---------------------------------------------------------------------------
#define BM 128
#define BN 128
#define BK 32

__global__ __launch_bounds__(256)
void gemm_bf16_kernel(const u16* __restrict__ A,   // [M,K] row-major
                      const u16* __restrict__ BT,  // [N,K] row-major (pre-transposed)
                      const float* __restrict__ bias, const float* __restrict__ resid,
                      float* __restrict__ outF, u16* __restrict__ outB,
                      int M, int N, int K, int relu) {
    __shared__ __align__(32) u16 As[2][BM][BK];
    __shared__ __align__(32) u16 Bs[2][BN][BK];
    int tid  = threadIdx.x;
    int lane = tid & 31, wave = tid >> 5;
    int l15  = lane & 15, kh = lane >> 4;
    int wm   = wave & 3;        // 0..3 -> 32-row band
    int wn   = wave >> 2;       // 0..1 -> 64-col band
    int rowBase = blockIdx.y * BM;
    int colBase = blockIdx.x * BN;

    // staging: each thread copies one contiguous 16-elem (32B) chunk of A and of BT
    int sa_row = tid >> 1;           // 0..127
    int sa_col = (tid & 1) * 16;     // 0 or 16
    const u16* aPtr = A  + (size_t)(rowBase + sa_row) * K + sa_col;
    const u16* bPtr = BT + (size_t)(colBase + sa_row) * K + sa_col;

    f32x8 acc[2][4];
#pragma unroll
    for (int mi = 0; mi < 2; ++mi)
#pragma unroll
        for (int ni = 0; ni < 4; ++ni)
            acc[mi][ni] = (f32x8){0, 0, 0, 0, 0, 0, 0, 0};

    int nsteps = K / BK;

    // prologue: stage tile 0
    bf16x16 aReg = *(const bf16x16*)aPtr;
    bf16x16 bReg = *(const bf16x16*)bPtr;
    *(bf16x16*)&As[0][sa_row][sa_col] = aReg;
    *(bf16x16*)&Bs[0][sa_row][sa_col] = bReg;
    __syncthreads();

    for (int step = 0; step < nsteps; ++step) {
        int cur = step & 1;
        if (step + 1 < nsteps) {
            aReg = *(const bf16x16*)(aPtr + (size_t)(step + 1) * BK);
            bReg = *(const bf16x16*)(bPtr + (size_t)(step + 1) * BK);
            if (step + 2 < nsteps) {
                __builtin_prefetch(aPtr + (size_t)(step + 2) * BK, 0, 0);
                __builtin_prefetch(bPtr + (size_t)(step + 2) * BK, 0, 0);
            }
        }
        // fragments (contiguous k order: lane-half selects K half for both A and B)
        ABFrag af[2];
#pragma unroll
        for (int mi = 0; mi < 2; ++mi) {
            int r = wm * 32 + mi * 16 + l15;
            af[mi].h[0] = *(const bf16x8*)&As[cur][r][kh * 16];
            af[mi].h[1] = *(const bf16x8*)&As[cur][r][kh * 16 + 8];
        }
#pragma unroll
        for (int ni = 0; ni < 4; ++ni) {
            int n = wn * 64 + ni * 16 + l15;
            bf16x16 bf = *(const bf16x16*)&Bs[cur][n][kh * 16];
            acc[0][ni] = __builtin_amdgcn_wmma_f32_16x16x32_bf16(false, af[0].v, false, bf,
                                                                 (short)0, acc[0][ni], false, false);
            acc[1][ni] = __builtin_amdgcn_wmma_f32_16x16x32_bf16(false, af[1].v, false, bf,
                                                                 (short)0, acc[1][ni], false, false);
        }
        if (step + 1 < nsteps) {
            *(bf16x16*)&As[cur ^ 1][sa_row][sa_col] = aReg;
            *(bf16x16*)&Bs[cur ^ 1][sa_row][sa_col] = bReg;
        }
        __syncthreads();
    }

#pragma unroll
    for (int mi = 0; mi < 2; ++mi)
#pragma unroll
        for (int ni = 0; ni < 4; ++ni)
#pragma unroll
            for (int r = 0; r < 8; ++r) {
                int row = rowBase + wm * 32 + mi * 16 + r + 8 * kh;
                int col = colBase + wn * 64 + ni * 16 + l15;
                float v0 = acc[mi][ni][r];
                if (bias)  v0 += bias[col];
                if (resid) v0 += resid[(size_t)row * N + col];
                if (relu)  v0 = fmaxf(v0, 0.0f);
                if (outF)  outF[(size_t)row * N + col] = v0;
                if (outB)  outB[(size_t)row * N + col] = f2bf(v0);
            }
}

// ---------------------------------------------------------------------------
// Flash attention w/ ALiBi + causal mask. One wave per 16 query rows.
// q,k: bf16 [B,T,H*HS]; vT: bf16 [B,H,HS,T]; o: bf16 [B,T,H*HS]
// ---------------------------------------------------------------------------
__global__ __launch_bounds__(256)
void attn_kernel(const u16* __restrict__ qb, const u16* __restrict__ kb,
                 const u16* __restrict__ vT, u16* __restrict__ ob) {
    __shared__ __align__(32) u16 Plds[8][16 * 32];
    int tid  = threadIdx.x;
    int lane = tid & 31, wave = tid >> 5;
    int l15  = lane & 15, kh = lane >> 4;
    int flat = blockIdx.x * 8 + wave;
    int qblk = flat % (T_ / 16);
    int bh   = flat / (T_ / 16);
    int b    = bh / H_, h = bh % H_;
    int q0   = qblk * 16;
    float slope = exp2f(-0.5f * (float)(h + 1));   // H=16 -> 2^(-8/16 * (h+1))
    const float scale = 0.125f;                    // HS^-0.5

    // Q fragments for rows t = q0 + l15, K = HS in two chunks of 32
    const u16* qrow = qb + (size_t)(b * T_ + q0 + l15) * C_ + h * HS_;
    ABFrag qf[2];
#pragma unroll
    for (int kc = 0; kc < 2; ++kc) {
        qf[kc].h[0] = *(const bf16x8*)(qrow + kc * 32 + kh * 16);
        qf[kc].h[1] = *(const bf16x8*)(qrow + kc * 32 + kh * 16 + 8);
    }

    f32x8 o[4];
#pragma unroll
    for (int dt = 0; dt < 4; ++dt) o[dt] = (f32x8){0, 0, 0, 0, 0, 0, 0, 0};
    float mrun[8], lrun[8];
#pragma unroll
    for (int r = 0; r < 8; ++r) { mrun[r] = -1e30f; lrun[r] = 0.0f; }

    const u16* kbase = kb + (size_t)b * T_ * C_ + h * HS_;
    const u16* vtb   = vT + (size_t)bh * HS_ * T_;
    int tmax = q0 + 15;

    for (int s0 = 0; s0 <= tmax; s0 += 32) {
        // ---- scores: two 16x16 tiles over this 32-wide key chunk
        f32x8 sc[2];
#pragma unroll
        for (int nt = 0; nt < 2; ++nt) {
            f32x8 a = {0, 0, 0, 0, 0, 0, 0, 0};
            const u16* kr = kbase + (size_t)(s0 + nt * 16 + l15) * C_;
#pragma unroll
            for (int kc = 0; kc < 2; ++kc) {
                bf16x16 bf = *(const bf16x16*)(kr + kc * 32 + kh * 16);
                a = __builtin_amdgcn_wmma_f32_16x16x32_bf16(false, qf[kc].v, false, bf,
                                                            (short)0, a, false, false);
            }
            sc[nt] = a;
        }
        // ---- ALiBi + causal mask + row max (reduce across 16 lanes)
        float mnew[8];
#pragma unroll
        for (int r = 0; r < 8; ++r) {
            int t = q0 + r + 8 * kh;
#pragma unroll
            for (int nt = 0; nt < 2; ++nt) {
                int s = s0 + nt * 16 + l15;
                float val = sc[nt][r] * scale - slope * fabsf((float)(t - s));
                if (s > t) val = -1e30f;
                sc[nt][r] = val;
            }
            float m = fmaxf(sc[0][r], sc[1][r]);
#pragma unroll
            for (int off = 1; off < 16; off <<= 1)
                m = fmaxf(m, __shfl_xor(m, off, 16));
            mnew[r] = fmaxf(mrun[r], m);
        }
        // ---- exp, row sums, rescale O, spill P tile to LDS (C-layout)
#pragma unroll
        for (int r = 0; r < 8; ++r) {
            int mrow = r + 8 * kh;
            float rs = 0.0f;
#pragma unroll
            for (int nt = 0; nt < 2; ++nt) {
                float p = __expf(sc[nt][r] - mnew[r]);
                rs += p;
                Plds[wave][mrow * 32 + nt * 16 + l15] = f2bf(p);
            }
#pragma unroll
            for (int off = 1; off < 16; off <<= 1)
                rs += __shfl_xor(rs, off, 16);
            float fr = __expf(mrun[r] - mnew[r]);
            lrun[r] = lrun[r] * fr + rs;
            mrun[r] = mnew[r];
#pragma unroll
            for (int dt = 0; dt < 4; ++dt) o[dt][r] *= fr;
        }
        asm volatile("s_wait_dscnt 0" ::: "memory");   // wave-local LDS RAW fence
        // ---- reload P in A-layout, accumulate P @ V into O
        ABFrag pf;
        pf.h[0] = *(const bf16x8*)&Plds[wave][l15 * 32 + kh * 16];
        pf.h[1] = *(const bf16x8*)&Plds[wave][l15 * 32 + kh * 16 + 8];
#pragma unroll
        for (int dt = 0; dt < 4; ++dt) {
            const u16* vr = vtb + (size_t)(dt * 16 + l15) * T_ + s0 + kh * 16;
            bf16x16 bv = *(const bf16x16*)vr;
            o[dt] = __builtin_amdgcn_wmma_f32_16x16x32_bf16(false, pf.v, false, bv,
                                                            (short)0, o[dt], false, false);
        }
    }
    // ---- normalize and store
#pragma unroll
    for (int r = 0; r < 8; ++r) {
        int t = q0 + r + 8 * kh;
        float inv = 1.0f / lrun[r];
        u16* orow = ob + (size_t)(b * T_ + t) * C_ + h * HS_;
#pragma unroll
        for (int dt = 0; dt < 4; ++dt)
            orow[dt * 16 + l15] = f2bf(o[dt][r] * inv);
    }
}

// ---------------------------------------------------------------------------
// Orchestration
// ---------------------------------------------------------------------------
extern "C" void kernel_launch(void* const* d_in, const int* in_sizes, int n_in,
                              void* d_out, int out_size, void* d_ws, size_t ws_size,
                              hipStream_t stream) {
    const float* x     = (const float*)d_in[0];
    const float* Wq    = (const float*)d_in[1];
    const float* Wk    = (const float*)d_in[2];
    const float* Wv    = (const float*)d_in[3];
    const float* Wproj = (const float*)d_in[4];
    const float* bproj = (const float*)d_in[5];
    const float* ln1g  = (const float*)d_in[6];
    const float* ln1b  = (const float*)d_in[7];
    const float* ln2g  = (const float*)d_in[8];
    const float* ln2b  = (const float*)d_in[9];
    const float* W1    = (const float*)d_in[10];
    const float* b1    = (const float*)d_in[11];
    const float* W2    = (const float*)d_in[12];
    const float* b2    = (const float*)d_in[13];
    float* out = (float*)d_out;

    char* ws = (char*)d_ws;
    const size_t MB = 1024 * 1024;
    u16*  qB   = (u16*)(ws + 0 * MB);     // 8MB
    u16*  kB   = (u16*)(ws + 8 * MB);     // 8MB
    u16*  vB   = (u16*)(ws + 16 * MB);    // 8MB
    u16*  vTB  = (u16*)(ws + 24 * MB);    // 8MB
    u16*  ffn1 = (u16*)(ws + 0 * MB);     // 32MB (overlays q/k/v/vT after attention)
    u16*  hB   = (u16*)(ws + 32 * MB);    // 8MB (h1, later h2)
    u16*  WqT  = (u16*)(ws + 40 * MB);    // 2MB
    u16*  WkT  = (u16*)(ws + 42 * MB);    // 2MB
    u16*  WvT  = (u16*)(ws + 44 * MB);    // 2MB
    u16*  WpT  = (u16*)(ws + 46 * MB);    // 2MB
    u16*  W1T  = (u16*)(ws + 48 * MB);    // 8MB
    u16*  W2T  = (u16*)(ws + 56 * MB);    // 8MB
    u16*  oB   = (u16*)(ws + 64 * MB);    // 8MB
    float* x1  = (float*)(ws + 72 * MB);  // 16MB -> 88MB total

    int nW = H_ * C_ * HS_;
    wqkvT_kernel<<<(nW + 255) / 256, 256, 0, stream>>>(Wq, WqT);
    wqkvT_kernel<<<(nW + 255) / 256, 256, 0, stream>>>(Wk, WkT);
    wqkvT_kernel<<<(nW + 255) / 256, 256, 0, stream>>>(Wv, WvT);
    cvtT_kernel<<<(C_ * C_ + 255) / 256, 256, 0, stream>>>(Wproj, WpT, C_, C_);
    cvtT_kernel<<<(C_ * 4 * C_ + 255) / 256, 256, 0, stream>>>(W1, W1T, C_, 4 * C_);
    cvtT_kernel<<<(C_ * 4 * C_ + 255) / 256, 256, 0, stream>>>(W2, W2T, 4 * C_, C_);

    ln_kernel<<<NTOK, 256, 0, stream>>>(x, ln1g, ln1b, hB);

    dim3 gQKV(C_ / BN, NTOK / BM);
    gemm_bf16_kernel<<<gQKV, 256, 0, stream>>>(hB, WqT, nullptr, nullptr, nullptr, qB, NTOK, C_, C_, 0);
    gemm_bf16_kernel<<<gQKV, 256, 0, stream>>>(hB, WkT, nullptr, nullptr, nullptr, kB, NTOK, C_, C_, 0);
    gemm_bf16_kernel<<<gQKV, 256, 0, stream>>>(hB, WvT, nullptr, nullptr, nullptr, vB, NTOK, C_, C_, 0);

    vtrans_kernel<<<(B_ * T_ * C_ + 255) / 256, 256, 0, stream>>>(vB, vTB);

    attn_kernel<<<(B_ * H_ * (T_ / 16)) / 8, 256, 0, stream>>>(qB, kB, vTB, oB);

    // x1 = x + o @ Wproj + bproj
    gemm_bf16_kernel<<<gQKV, 256, 0, stream>>>(oB, WpT, bproj, x, x1, nullptr, NTOK, C_, C_, 0);

    ln_kernel<<<NTOK, 256, 0, stream>>>(x1, ln2g, ln2b, hB);

    dim3 gF1(4 * C_ / BN, NTOK / BM);
    gemm_bf16_kernel<<<gF1, 256, 0, stream>>>(hB, W1T, b1, nullptr, nullptr, ffn1, NTOK, 4 * C_, C_, 1);

    dim3 gF2(C_ / BN, NTOK / BM);
    gemm_bf16_kernel<<<gF2, 256, 0, stream>>>(ffn1, W2T, b2, x1, out, nullptr, NTOK, C_, 4 * C_, 0);
}
